// Wcompute_11450382811365
// MI455X (gfx1250) — compile-verified, matrix operand order checked by reading, not run on hardware
//
#include <hip/hip_runtime.h>
#include <hip/hip_bf16.h>
#include <stdint.h>

// W-compute graph head for MI455X (gfx1250), wave32 + v_wmma_f32_16x16x32_bf16.
//
// Optimizations (reasoned from MI455X specs, compile-only):
//  * w1t (96KB bf16) staged in LDS once per workgroup (fits 320KB WGP LDS),
//    async-to-LDS when the builtin exists; B fragments then read as
//    ds_load_b128 -> B-operand L2 traffic cut 8x.
//  * (i,j) symmetry of diff/h/s: stats pass weights j>i by 2 and skips
//    sub-diagonal tiles; score pass computes j>=i only and mirrors the
//    write -> total WMMA work halved (~13 GFLOP).
//  * diff f32->bf16 pair-pack in ONE VALU op: v_cvt_pk_bf16_f32 if available,
//    else v_perm_b32 (truncating pack of the two high halves). Weights (one
//    time) keep RNE.
//
// WMMA fragment layouts (CDNA5 ISA):
//  * A (16x32 bf16): lane l: row m=l&15, half=l>>4;
//      vgpr v -> K = (v<4 ? 0 : 16) + half*8 + 2*(v&3), pair (K, K+1).
//  * B (32x16 bf16): mirrored (lane = column n=l&15, same K map) -- each
//    lane's 8 dwords = two contiguous 16B blocks of w1t[c][*].
//  * C/D f32: vgpr r, lane l -> M = r + 8*(l>>4), N = l&15.

typedef uint32_t u32;
typedef __attribute__((ext_vector_type(16))) __bf16 v16bf;
typedef __attribute__((ext_vector_type(8)))  float  v8f;
typedef __attribute__((ext_vector_type(8)))  u32    v8u;

#define BB 4
#define NN 256
#define FF 256
#define NF 192
#define NTILES 12   // NF/16
#define KCH 8       // FF/32 K-steps of 32
#define W1_BYTES (NF * FF * 2)        // 98304
#define W1_VEC   (W1_BYTES / 16)      // 6144 uint4s

// Pointer types matching the async-to-LDS builtin's signature (AS1/AS3 int4*).
typedef int v4i __attribute__((vector_size(16)));
typedef __attribute__((address_space(1))) v4i g_v4i;
typedef __attribute__((address_space(3))) v4i lds_v4i;

__device__ __forceinline__ u32 f2bf(float f) {          // RNE f32 -> bf16 bits
  u32 u = __float_as_uint(f);
  return (u + 0x7fffu + ((u >> 16) & 1u)) >> 16;
}

#if __has_builtin(__builtin_amdgcn_cvt_pk_bf16_f32)
typedef __attribute__((ext_vector_type(2))) __bf16 v2bf;
__device__ __forceinline__ u32 pack2(float a, float b) { // 1 op, RNE
  return __builtin_bit_cast(u32, __builtin_amdgcn_cvt_pk_bf16_f32(a, b));
}
#elif __has_builtin(__builtin_amdgcn_perm)
__device__ __forceinline__ u32 pack2(float a, float b) { // 1 op, truncating:
  // v_perm_b32: bytes 0-3 select from s1(=a), 4-7 from s0(=b);
  // result = {b[31:24], b[23:16], a[31:24], a[23:16]} = {bf16(b), bf16(a)}
  return __builtin_amdgcn_perm(__float_as_uint(b), __float_as_uint(a),
                               0x07060302u);
}
#else
__device__ __forceinline__ u32 pack2(float a, float b) {
  return f2bf(a) | (f2bf(b) << 16);
}
#endif

__device__ __forceinline__ int koff(int v, int half) {  // K offset within 32-chunk
  int lo = half * 8 + ((v & 3) << 1);
  return (v < 4) ? lo : lo + 16;
}
__device__ __forceinline__ v16bf as_bf16x16(const u32* p) {
  v8u t;
#pragma unroll
  for (int k = 0; k < 8; ++k) t[k] = p[k];
  return __builtin_bit_cast(v16bf, t);
}
__device__ __forceinline__ v16bf as_bf16x16_2(uint4 a, uint4 b) {
  v8u t;
  t[0] = a.x; t[1] = a.y; t[2] = a.z; t[3] = a.w;
  t[4] = b.x; t[5] = b.y; t[6] = b.z; t[7] = b.w;
  return __builtin_bit_cast(v16bf, t);
}

// Stage w1t (bf16 [NF][FF], 96KB) from global into LDS, whole workgroup.
__device__ __forceinline__ void stage_w1(const uint4* __restrict__ g,
                                         uint4* s, int tid) {
#if __has_builtin(__builtin_amdgcn_global_load_async_to_lds_b128)
#pragma unroll 1
  for (int n = tid; n < W1_VEC; n += 256)
    __builtin_amdgcn_global_load_async_to_lds_b128(
        (g_v4i*)(g + n), (lds_v4i*)(s + n), 0, 0);
#if __has_builtin(__builtin_amdgcn_s_wait_asynccnt)
  __builtin_amdgcn_s_wait_asynccnt(0);
#else
  asm volatile("s_wait_asynccnt 0x0" ::: "memory");
#endif
#else
#pragma unroll 1
  for (int n = tid; n < W1_VEC; n += 256) s[n] = g[n];
#endif
}

// Build A fragments for all 8 K-chunks: row m of the tile is |x_i - x_{j0+m}|.
__device__ __forceinline__ void build_a(const float* __restrict__ xi,
                                        const float* __restrict__ xj,
                                        int half, u32 afr[KCH][8]) {
#pragma unroll
  for (int kc = 0; kc < KCH; ++kc) {
#pragma unroll
    for (int v = 0; v < 8; ++v) {
      int k = kc * 32 + koff(v, half);
      float d0 = fabsf(xi[k]     - xj[k]);
      float d1 = fabsf(xi[k + 1] - xj[k + 1]);
      afr[kc][v] = pack2(d0, d1);
    }
  }
}

// One 16x16 tile: D = sum_K A(diff) x B(w1 block), B fragments from LDS.
// sw_c points at this lane's channel row: (uint4*)w1s + c*32.
__device__ __forceinline__ v8f mm_tile(const u32 afr[KCH][8],
                                       const uint4* __restrict__ sw_c, int half) {
  v8f acc = {0.f, 0.f, 0.f, 0.f, 0.f, 0.f, 0.f, 0.f};
#pragma unroll
  for (int kc = 0; kc < KCH; ++kc) {
    uint4 b0 = sw_c[kc * 4 + half];      // ds_load_b128
    uint4 b1 = sw_c[kc * 4 + half + 2];  // ds_load_b128
    acc = __builtin_amdgcn_wmma_f32_16x16x32_bf16(
        false, as_bf16x16(&afr[kc][0]),
        false, as_bf16x16_2(b0, b1),
        (short)0, acc, false, false);
  }
  return acc;
}

// ---- kernel 0: w1 (f32 [F][NF]) -> w1t (bf16 [NF][F]); zero stats[0..255] ----
__global__ __launch_bounds__(256) void k_prep(const float* __restrict__ w1,
                                              unsigned short* __restrict__ w1t,
                                              float* __restrict__ stats) {
  if (blockIdx.x == 0 && threadIdx.x < 2 * NF) stats[threadIdx.x] = 0.f;
  int idx = blockIdx.x * 256 + threadIdx.x;   // 192*256 = NF*FF exactly
  int c = idx >> 8, f = idx & 255;
  w1t[c * FF + f] = (unsigned short)f2bf(w1[f * NF + c]);
}

// ---- kernel 1: per-channel sum / sumsq of D (stats pass, upper triangle) ----
__global__ __launch_bounds__(256) void k_stats(const float* __restrict__ x,
                                               const unsigned short* __restrict__ w1t,
                                               float* __restrict__ stats) {
  extern __shared__ unsigned char dynsmem[];
  uint4* w1s  = (uint4*)dynsmem;
  float* sred = (float*)(dynsmem + W1_BYTES);

  int tid = threadIdx.x;
  stage_w1((const uint4*)w1t, w1s, tid);
  for (int idx = tid; idx < 2 * NF; idx += 256) sred[idx] = 0.f;
  __syncthreads();

  int wid = tid >> 5, lane = tid & 31, half = lane >> 4, m = lane & 15;
  int t  = blockIdx.x * 8 + wid;          // 16384 M-tiles
  int b  = t >> 12, tb = t & 4095;
  int i  = tb >> 4, j0 = (tb & 15) << 4;  // tile rows = (b, i, j0..j0+15)

  if (j0 + 15 > i) {                       // symmetry: only tiles with some j>i
    const float* xb = x + (size_t)b * NN * FF;
    const float* xi = xb + (size_t)i * FF;
    const float* xj = xb + (size_t)(j0 + m) * FF;
    u32 afr[KCH][8];
    build_a(xi, xj, half, afr);

    float wgt[8];
#pragma unroll
    for (int r = 0; r < 8; ++r) {          // j>i counted twice; j<=i zero (D(i,i)=0)
      int j = j0 + r + 8 * half;
      wgt[r] = (j > i) ? 2.f : 0.f;
    }

#pragma unroll 1
    for (int nt = 0; nt < NTILES; ++nt) {
      int c = nt * 16 + m;                 // this lane's channel
      v8f d = mm_tile(afr, (const uint4*)w1s + c * 32, half);
      float ls = 0.f, lq = 0.f;
#pragma unroll
      for (int r = 0; r < 8; ++r) { ls += wgt[r] * d[r]; lq += wgt[r] * d[r] * d[r]; }
      ls += __shfl_xor(ls, 16, 32);        // lanes l and l+16 share channel c
      lq += __shfl_xor(lq, 16, 32);
      if (half == 0) {
        atomicAdd(&sred[c], ls);
        atomicAdd(&sred[NF + c], lq);
      }
    }
  }
  __syncthreads();
  for (int idx = tid; idx < 2 * NF; idx += 256)
    atomicAdd(&stats[idx], sred[idx]);
}

// ---- kernel 2: BN fold: scale = g*rsqrt(var+eps), shift = b - mean*scale ----
__global__ void k_finalize(const float* __restrict__ stats,
                           const float* __restrict__ gamma,
                           const float* __restrict__ beta,
                           float* __restrict__ ss) {
  int c = threadIdx.x;                    // 192 threads
  const float invM = 1.0f / (float)(BB * NN * NN);
  float mean = stats[c] * invM;
  float var  = fmaxf(stats[NF + c] * invM - mean * mean, 0.f);
  float sc   = gamma[c] * rsqrtf(var + 1e-5f);
  ss[c]      = sc;
  ss[NF + c] = beta[c] - mean * sc;
}

// ---- kernel 3: D -> BN + leaky + dot(w2) -> s[b,i,j], j>=i + mirror ----
__global__ __launch_bounds__(256) void k_score(const float* __restrict__ x,
                                               const unsigned short* __restrict__ w1t,
                                               const float* __restrict__ ss,
                                               const float* __restrict__ w2,
                                               float* __restrict__ sbuf) {
  extern __shared__ unsigned char dynsmem[];
  uint4* w1s = (uint4*)dynsmem;

  int tid = threadIdx.x;
  stage_w1((const uint4*)w1t, w1s, tid);
  __syncthreads();

  int wid = tid >> 5, lane = tid & 31, half = lane >> 4, m = lane & 15;
  int t  = blockIdx.x * 8 + wid;
  int b  = t >> 12, tb = t & 4095;
  int i  = tb >> 4, j0 = (tb & 15) << 4;

  if (j0 + 15 < i) return;                 // symmetry: tile entirely below diag

  const float* xb = x + (size_t)b * NN * FF;
  const float* xi = xb + (size_t)i * FF;
  const float* xj = xb + (size_t)(j0 + m) * FF;
  u32 afr[KCH][8];
  build_a(xi, xj, half, afr);

  float part[8];
#pragma unroll
  for (int r = 0; r < 8; ++r) part[r] = 0.f;

#pragma unroll 1
  for (int nt = 0; nt < NTILES; ++nt) {
    int c = nt * 16 + m;
    v8f d = mm_tile(afr, (const uint4*)w1s + c * 32, half);
    float sc = ss[c], sh = ss[NF + c], w = w2[c];
#pragma unroll
    for (int r = 0; r < 8; ++r) {
      float h = d[r] * sc + sh;           // BatchNorm (b1 cancels)
      h = (h >= 0.f) ? h : 0.01f * h;     // leaky_relu
      part[r] += h * w;                   // contract channels
    }
  }
  // sum over the 16 channel-lanes in each half (rows M = r + 8*half)
#pragma unroll
  for (int r = 0; r < 8; ++r) {
#pragma unroll
    for (int msk = 1; msk <= 8; msk <<= 1)
      part[r] += __shfl_xor(part[r], msk, 32);
  }
  if (m == 0) {
    float* sb = sbuf + (size_t)b * NN * NN;
#pragma unroll
    for (int r = 0; r < 8; ++r) {
      int j = j0 + r + 8 * half;
      float sv = part[r];                 // b2 cancels in softmax
      if (j == i) sb[i * NN + j] = sv - 1e8f;      // masked diagonal
      else if (j > i) {
        sb[i * NN + j] = sv;                       // upper
        sb[j * NN + i] = sv;                       // mirror (s symmetric)
      }
    }
  }
}

// ---- kernel 4: row softmax over j, out = eye - A ----
__global__ __launch_bounds__(256) void k_softmax(const float* __restrict__ s,
                                                 float* __restrict__ out) {
  int row  = blockIdx.x * 8 + (threadIdx.x >> 5);  // 1024 rows (b,i)
  int lane = threadIdx.x & 31;
  int i    = row & (NN - 1);
  const float* sr = s + (size_t)row * NN;
  float v[8];
  float mx = -3.0e38f;
#pragma unroll
  for (int k = 0; k < 8; ++k) { v[k] = sr[lane + 32 * k]; mx = fmaxf(mx, v[k]); }
#pragma unroll
  for (int msk = 16; msk >= 1; msk >>= 1) mx = fmaxf(mx, __shfl_xor(mx, msk, 32));
  float sum = 0.f;
#pragma unroll
  for (int k = 0; k < 8; ++k) { v[k] = __expf(v[k] - mx); sum += v[k]; }
#pragma unroll
  for (int msk = 16; msk >= 1; msk >>= 1) sum += __shfl_xor(sum, msk, 32);
  float inv = 1.f / sum;
#pragma unroll
  for (int k = 0; k < 8; ++k) {
    int j = lane + 32 * k;
    out[(size_t)row * NN + j] = ((j == i) ? 1.f : 0.f) - v[k] * inv;
  }
}

extern "C" void kernel_launch(void* const* d_in, const int* in_sizes, int n_in,
                              void* d_out, int out_size, void* d_ws, size_t ws_size,
                              hipStream_t stream) {
  // setup_inputs order: x, W_id, w1, b1, gamma, beta, w2, b2
  const float* x     = (const float*)d_in[0];
  const float* w1    = (const float*)d_in[2];
  const float* gamma = (const float*)d_in[4];
  const float* beta  = (const float*)d_in[5];
  const float* w2    = (const float*)d_in[6];
  float* out = (float*)d_out;

  char* ws = (char*)d_ws;
  unsigned short* w1t = (unsigned short*)ws;            // [NF][FF] bf16: 98304 B
  float* stats = (float*)(ws + 98304);                  // sum[192] + sumsq[192]
  float* ss    = (float*)(ws + 99840);                  // scale[192] + shift[192]
  float* sbuf  = (float*)(ws + 101376);                 // s[B][N][N]: 1 MB

  k_prep    <<<192,  256, 0, stream>>>(w1, w1t, stats);
  k_stats   <<<2048, 256, W1_BYTES + 2 * NF * 4, stream>>>(x, w1t, stats);
  k_finalize<<<1,    192, 0, stream>>>(stats, gamma, beta, ss);
  k_score   <<<2048, 256, W1_BYTES, stream>>>(x, w1t, ss, w2, sbuf);
  k_softmax <<<128,  256, 0, stream>>>(sbuf, out);
}
